// RelativePositionalEncoding_74182675136571
// MI455X (gfx1250) — matrix operand order repaired and checked by reading, not compiled, not run on hardware
//
#include <hip/hip_runtime.h>
#include <hip/hip_bf16.h>

// ---------------------------------------------------------------------------
// RelativePositionalEncoding on MI455X (gfx1250, wave32, WMMA)
//
// out[b,i,:] = x[b,i,:] + (1/S) * counts[b,i,:] @ emb_table      (counts is a
// per-row histogram of clamp(pos_i - pos_j, -M, M) + M over all j)
//
// Strategy: fuse histogram (LDS, ds_add_f32 atomics) with an FP16 WMMA GEMM
// (counts exact in f16 since <= 2048). Table is pre-converted/transposed to
// f16 [D][Kpad] in d_ws so B-fragments are contiguous 32B loads from L2.
// ---------------------------------------------------------------------------

typedef __attribute__((ext_vector_type(16))) _Float16 v16h;
typedef __attribute__((ext_vector_type(8)))  float    v8f;
typedef __attribute__((ext_vector_type(4)))  float    v4f;

#define B_SZ   4
#define S_SZ   2048
#define D_SZ   512
#define M_CLAMP 2048
#define V_SZ   4097           // 2*M+1
#define KPAD   4128           // V padded to a multiple of 32
#define MROWS  16             // query rows per workgroup
#define NTHREADS 256          // 8 wave32s

// ---------------------------------------------------------------------------
// Kernel 1: emb_table (f32, [V][D]) -> tT (f16, [D][KPAD]), transposed and
// K-zero-padded so padded K-tiles contribute exactly zero.
// ---------------------------------------------------------------------------
__global__ void __launch_bounds__(256)
relpos_cvt_table(const float* __restrict__ emb, _Float16* __restrict__ tT) {
    int k = blockIdx.x * 256 + threadIdx.x;
    if (k >= KPAD) return;
    if (k < V_SZ) {
        const float* src = emb + (size_t)k * D_SZ;
        for (int d = 0; d < D_SZ; ++d)
            tT[(size_t)d * KPAD + k] = (_Float16)src[d];
    } else {
        for (int d = 0; d < D_SZ; ++d)
            tT[(size_t)d * KPAD + k] = (_Float16)0.0f;
    }
}

// ---------------------------------------------------------------------------
// Kernel 2: fused histogram + WMMA GEMM + epilogue.
// One WG = 16 consecutive rows of one batch, all 512 output columns.
// Dynamic LDS: float cnt[16][KPAD] (264192 B) + int posLDS[2048] (8192 B)
//            = 272384 B  (CDNA5 allows up to 320 KB per workgroup)
// ---------------------------------------------------------------------------
__global__ void __launch_bounds__(NTHREADS)
relpos_main(const float* __restrict__ x,
            const _Float16* __restrict__ tT,
            const int* __restrict__ pos,
            float* __restrict__ out) {
    extern __shared__ char smem[];
    float* cnt    = (float*)smem;                                // [MROWS][KPAD]
    int*   posLDS = (int*)(smem + (size_t)MROWS * KPAD * 4);     // [S_SZ]

    const int tid = threadIdx.x;
    const int wg  = blockIdx.x;
    const int b   = (wg * MROWS) / S_SZ;
    const int i0  = (wg * MROWS) % S_SZ;

    // ---- zero counts, stage this batch's positions ----
    for (int idx = tid; idx < MROWS * KPAD; idx += NTHREADS) cnt[idx] = 0.0f;
    for (int j = tid; j < S_SZ; j += NTHREADS) posLDS[j] = pos[b * S_SZ + j];
    __syncthreads();

    // ---- histogram: 16 threads per row, ds_add_f32 atomics (exact) ----
    {
        const int row = tid & (MROWS - 1);
        const int p_i = posLDS[i0 + row];
        float* crow = cnt + row * KPAD;
        for (int j = tid >> 4; j < S_SZ; j += NTHREADS / MROWS) {
            int d = p_i - posLDS[j];
            d = d < -M_CLAMP ? -M_CLAMP : (d > M_CLAMP ? M_CLAMP : d);
            atomicAdd(&crow[d + M_CLAMP], 1.0f);
        }
    }
    __syncthreads();

    // ---- active K range (uniform across WG; positions are sorted) ----
    int vmin = posLDS[i0] - posLDS[S_SZ - 1];            // most negative diff
    if (vmin < -M_CLAMP) vmin = -M_CLAMP;
    vmin += M_CLAMP;
    int vmax = posLDS[i0 + MROWS - 1] - posLDS[0];       // most positive diff
    if (vmax > M_CLAMP) vmax = M_CLAMP;
    vmax += M_CLAMP;
    const int kbeg = (vmin >> 5) << 5;                   // 32-aligned
    const int kend = vmax + 1;

    // ---- WMMA GEMM: wave w owns columns [w*64, w*64+64) ----
    const int lane   = tid & 31;
    const int wave   = tid >> 5;
    const int laneHi = lane >> 4;        // 0: lanes 0-15, 1: lanes 16-31
    const int MN     = lane & 15;        // row (A/C) and column (B/C) index

    v8f acc[4] = {v8f{}, v8f{}, v8f{}, v8f{}};

    const float*    arow  = cnt + (size_t)MN * KPAD;
    const _Float16* bcol0 = tT + (size_t)(wave * 64 + MN) * KPAD;

    for (int k0 = kbeg; k0 < kend; k0 += 32) {
        // A fragment (16x32 f16): lane<16 holds K {0..15}+k0 of row MN,
        // lane>=16 holds K {16..31}+k0.  One 16-float run from LDS.
        v16h a;
        {
            const float* ap = arow + k0 + laneHi * 16;
            v4f f0 = *(const v4f*)(ap + 0);
            v4f f1 = *(const v4f*)(ap + 4);
            v4f f2 = *(const v4f*)(ap + 8);
            v4f f3 = *(const v4f*)(ap + 12);
#pragma unroll
            for (int t = 0; t < 4; ++t) {
                a[t]      = (_Float16)f0[t];
                a[4 + t]  = (_Float16)f1[t];
                a[8 + t]  = (_Float16)f2[t];
                a[12 + t] = (_Float16)f3[t];
            }
        }
        // B fragments (32x16 f16): lane holds N = MN, K = laneHi*16 + 0..15
        // -> 16 contiguous f16 (32B, aligned) in the transposed table.
#pragma unroll
        for (int t = 0; t < 4; ++t) {
            const _Float16* bp = bcol0 + (size_t)t * 16 * KPAD + k0 + laneHi * 16;
            v16h bf = *(const v16h*)bp;
            acc[t] = __builtin_amdgcn_wmma_f32_16x16x32_f16(
                false, a, false, bf, (short)0, acc[t], false, false);
        }
    }

    // ---- epilogue: out = x + acc / S ----
    const float invS = 1.0f / (float)S_SZ;
    const size_t rowBase = (size_t)b * S_SZ * D_SZ;
#pragma unroll
    for (int t = 0; t < 4; ++t) {
        const int col = wave * 64 + t * 16 + MN;
#pragma unroll
        for (int r = 0; r < 8; ++r) {
            const int Mi = r + laneHi * 8;                 // C layout: VGPR r
            const size_t idx = rowBase + (size_t)(i0 + Mi) * D_SZ + col;
            out[idx] = x[idx] + acc[t][r] * invS;
        }
    }
}

// ---------------------------------------------------------------------------
extern "C" void kernel_launch(void* const* d_in, const int* in_sizes, int n_in,
                              void* d_out, int out_size, void* d_ws, size_t ws_size,
                              hipStream_t stream) {
    const float* x   = (const float*)d_in[0];  // [B,S,D] f32
    const float* emb = (const float*)d_in[1];  // [V,D]   f32
    const int*   pos = (const int*)d_in[2];    // [B,S]   i32
    float*       out = (float*)d_out;          // [B,S,D] f32
    _Float16*    tT  = (_Float16*)d_ws;        // [D][KPAD] f16 (4.2 MB)

    relpos_cvt_table<<<(KPAD + 255) / 256, 256, 0, stream>>>(emb, tT);

    const size_t ldsBytes = (size_t)MROWS * KPAD * 4 + (size_t)S_SZ * 4; // 272384
    const int nBlocks = (B_SZ * S_SZ) / MROWS;                            // 512
    relpos_main<<<nBlocks, NTHREADS, ldsBytes, stream>>>(x, tT, pos, out);
}
